// MultiHeadAttention_27144193311055
// MI455X (gfx1250) — matrix operand order repaired and checked by reading, not compiled, not run on hardware
//
#include <hip/hip_runtime.h>
#include <hip/hip_bf16.h>

// ---------------------------------------------------------------------------
// MHA forward: B=4, T=2048, D=768, H=12, Dh=64.  bf16 WMMA, fp32 accumulate.
// GEMMs use 32x64 wave tiles; operands batched ahead of WMMA issue for ILP.
// ---------------------------------------------------------------------------

#define B_   4
#define T_   2048
#define D_   768
#define H_   12
#define DH_  64
#define M_   (B_ * T_)        // 8192 rows
#define D3_  (3 * D_)         // 2304

typedef __bf16 bf16;
typedef __attribute__((ext_vector_type(16))) bf16  v16bf;
typedef __attribute__((ext_vector_type(8)))  bf16  v8bf;
typedef __attribute__((ext_vector_type(8)))  float v8f;

union BF16x16 { v16bf v; v8bf h[2]; };

__device__ __forceinline__ v8bf load8(const bf16* p) {
    return *(const v8bf*)p;   // global_load_b128 (all call sites 16B aligned)
}

// ------------------------------ fp32 -> bf16 -------------------------------
__global__ void __launch_bounds__(256)
cvt_f32_to_bf16_kernel(const float* __restrict__ src, bf16* __restrict__ dst, int n) {
    int i = (blockIdx.x * 256 + threadIdx.x) * 4;
    if (i < n) {
        float4 f = *(const float4*)(src + i);
        dst[i + 0] = (bf16)f.x;
        dst[i + 1] = (bf16)f.y;
        dst[i + 2] = (bf16)f.z;
        dst[i + 3] = (bf16)f.w;
    }
}

// ------------------------------ QKV projection -----------------------------
// C[M, 3D] = xb[M,D] * Wqkv[3D,D]^T + bias ; scatter into Q/K/V^T bf16 tensors
// One wave computes a 32x64 tile of C.
__global__ void __launch_bounds__(256)
qkv_gemm_kernel(const bf16* __restrict__ xb, const bf16* __restrict__ wb,
                const float* __restrict__ bias,
                bf16* __restrict__ Qb, bf16* __restrict__ Kb, bf16* __restrict__ Vt) {
    const int lane = threadIdx.x & 31;
    const int wave = threadIdx.x >> 5;
    const int tilesN = D3_ / 64;                 // 36
    const int flat   = blockIdx.x * 8 + wave;    // wave-tile id
    const int tm = flat / tilesN;                // 0..255 (32-row tiles)
    const int tn = flat % tilesN;
    if (tm >= M_ / 32) return;
    const int m0 = tm * 32, n0 = tn * 64;
    const int l16 = lane & 15, lo = lane >> 4;

    v8f acc[2][4] = {};
    const bf16* pA0 = xb + (size_t)(m0 + l16) * D_ + lo * 8;
    const bf16* pA1 = pA0 + (size_t)16 * D_;

    for (int kk = 0; kk < D_; kk += 32) {
        BF16x16 a0, a1, bm[4];
        a0.h[0] = load8(pA0 + kk);  a0.h[1] = load8(pA0 + kk + 16);
        a1.h[0] = load8(pA1 + kk);  a1.h[1] = load8(pA1 + kk + 16);
#pragma unroll
        for (int c = 0; c < 4; ++c) {
            const bf16* pB = wb + (size_t)(n0 + c * 16 + l16) * D_ + kk + lo * 8;
            bm[c].h[0] = load8(pB);
            bm[c].h[1] = load8(pB + 16);
        }
#pragma unroll
        for (int c = 0; c < 4; ++c)
            acc[0][c] = __builtin_amdgcn_wmma_f32_16x16x32_bf16(
                false, a0.v, false, bm[c].v, (short)0, acc[0][c], false, false);
#pragma unroll
        for (int c = 0; c < 4; ++c)
            acc[1][c] = __builtin_amdgcn_wmma_f32_16x16x32_bf16(
                false, a1.v, false, bm[c].v, (short)0, acc[1][c], false, false);
    }

#pragma unroll
    for (int half = 0; half < 2; ++half)
#pragma unroll
        for (int c = 0; c < 4; ++c) {
            const int n = n0 + c * 16 + l16;
            const float bv = bias[n];
#pragma unroll
            for (int r = 0; r < 8; ++r) {
                const int m = m0 + half * 16 + r + lo * 8;
                const int bi = m / T_, t = m % T_;
                const float v = acc[half][c][r] + bv;
                if (n < D_) {                        // Q (scale folded in)
                    const int hh = n >> 6, dh = n & 63;
                    Qb[(((size_t)(bi * H_ + hh)) * T_ + t) * DH_ + dh] = (bf16)(v * 0.125f);
                } else if (n < 2 * D_) {             // K
                    const int n2 = n - D_;
                    const int hh = n2 >> 6, dh = n2 & 63;
                    Kb[(((size_t)(bi * H_ + hh)) * T_ + t) * DH_ + dh] = (bf16)v;
                } else {                             // V stored transposed [b,h,dh,t]
                    const int n2 = n - 2 * D_;
                    const int hh = n2 >> 6, dh = n2 & 63;
                    Vt[(((size_t)(bi * H_ + hh)) * DH_ + dh) * T_ + t] = (bf16)v;
                }
            }
        }
}

// --------------------------- flash attention -------------------------------
// one wave = one (b,h) 16-row Q tile; loop keys in steps of 32
__global__ void __launch_bounds__(256)
attn_kernel(const bf16* __restrict__ Qb, const bf16* __restrict__ Kb,
            const bf16* __restrict__ Vt, bf16* __restrict__ Ob) {
    __shared__ float pbuf[8][16][33];            // per-wave P transpose buffer

    const int lane = threadIdx.x & 31;
    const int wave = threadIdx.x >> 5;
    const int flat = blockIdx.x * 8 + wave;
    const int qtiles = T_ / 16;                  // 128
    const int qt = flat % qtiles;
    const int bh = flat / qtiles;                // 0..B*H-1
    if (bh >= B_ * H_) return;
    const int qbase = qt * 16;
    const int l16 = lane & 15, lo = lane >> 4;

    // Q tile, A layout, dh split into two K=32 chunks
    BF16x16 Aq[2];
    const bf16* pQ = Qb + ((size_t)bh * T_ + qbase + l16) * DH_ + lo * 8;
    Aq[0].h[0] = load8(pQ);       Aq[0].h[1] = load8(pQ + 16);
    Aq[1].h[0] = load8(pQ + 32);  Aq[1].h[1] = load8(pQ + 48);

    v8f o[4] = {};
    float mrow[8], lrow[8];
#pragma unroll
    for (int r = 0; r < 8; ++r) { mrow[r] = -1e30f; lrow[r] = 0.0f; }

    for (int j = 0; j < T_; j += 32) {
        // ---- batch all K-tile operand loads, then 4 S-WMMAs ----
        BF16x16 bk[2][2];
#pragma unroll
        for (int kh = 0; kh < 2; ++kh) {
            const bf16* pK = Kb + ((size_t)bh * T_ + j + kh * 16 + l16) * DH_ + lo * 8;
            bk[kh][0].h[0] = load8(pK);       bk[kh][0].h[1] = load8(pK + 16);   // dh 0..31
            bk[kh][1].h[0] = load8(pK + 32);  bk[kh][1].h[1] = load8(pK + 48);   // dh 32..63
        }
        v8f S[2];
#pragma unroll
        for (int kh = 0; kh < 2; ++kh) {
            v8f z = {};
            z     = __builtin_amdgcn_wmma_f32_16x16x32_bf16(
                        false, Aq[0].v, false, bk[kh][0].v, (short)0, z, false, false);
            S[kh] = __builtin_amdgcn_wmma_f32_16x16x32_bf16(
                        false, Aq[1].v, false, bk[kh][1].v, (short)0, z, false, false);
        }

        // ---- online softmax over the 32 new columns ----
        float mnew[8], alpha[8];
#pragma unroll
        for (int r = 0; r < 8; ++r) {
            float t0 = fmaxf(S[0][r], S[1][r]);
#pragma unroll
            for (int off = 8; off >= 1; off >>= 1)
                t0 = fmaxf(t0, __shfl_xor(t0, off, 16));
            mnew[r]  = fmaxf(mrow[r], t0);
            alpha[r] = __expf(mrow[r] - mnew[r]);
            mrow[r]  = mnew[r];
        }
#pragma unroll
        for (int r = 0; r < 8; ++r) {
            const float p0 = __expf(S[0][r] - mnew[r]);
            const float p1 = __expf(S[1][r] - mnew[r]);
            pbuf[wave][r + lo * 8][l16]      = p0;
            pbuf[wave][r + lo * 8][16 + l16] = p1;
            float rs = p0 + p1;
#pragma unroll
            for (int off = 8; off >= 1; off >>= 1)
                rs += __shfl_xor(rs, off, 16);
            lrow[r] = lrow[r] * alpha[r] + rs;
        }
#pragma unroll
        for (int c = 0; c < 4; ++c)
#pragma unroll
            for (int r = 0; r < 8; ++r) o[c][r] *= alpha[r];

        asm volatile("s_wait_dscnt 0" ::: "memory");

        // ---- read P back in WMMA A layout (16x32 bf16) ----
        BF16x16 Pa;
#pragma unroll
        for (int i = 0; i < 8; ++i) {
            Pa.v[i]     = (bf16)pbuf[wave][l16][lo * 8 + i];
            Pa.v[i + 8] = (bf16)pbuf[wave][l16][16 + lo * 8 + i];
        }

        // ---- batch all V-tile loads, then 4 PV-WMMAs ----
        BF16x16 bv[4];
#pragma unroll
        for (int c = 0; c < 4; ++c) {
            const bf16* pV = Vt + ((size_t)bh * DH_ + c * 16 + l16) * T_ + j + lo * 8;
            bv[c].h[0] = load8(pV);
            bv[c].h[1] = load8(pV + 16);
        }
#pragma unroll
        for (int c = 0; c < 4; ++c)
            o[c] = __builtin_amdgcn_wmma_f32_16x16x32_bf16(
                false, Pa.v, false, bv[c].v, (short)0, o[c], false, false);
    }

    // ---- epilogue: normalize and scatter to [B,T,D] bf16 ----
    const int bi = bh / H_, hh = bh % H_;
    float inv[8];
#pragma unroll
    for (int r = 0; r < 8; ++r) inv[r] = 1.0f / lrow[r];
#pragma unroll
    for (int c = 0; c < 4; ++c)
#pragma unroll
        for (int r = 0; r < 8; ++r) {
            const int t   = qbase + r + lo * 8;
            const int col = hh * DH_ + c * 16 + l16;
            Ob[((size_t)bi * T_ + t) * D_ + col] = (bf16)(o[c][r] * inv[r]);
        }
}

// ----------------------------- output projection ---------------------------
// One wave computes a 32x64 tile of out.
__global__ void __launch_bounds__(256)
proj_gemm_kernel(const bf16* __restrict__ ob, const bf16* __restrict__ wb,
                 const float* __restrict__ bias, float* __restrict__ out) {
    const int lane = threadIdx.x & 31;
    const int wave = threadIdx.x >> 5;
    const int tilesN = D_ / 64;                  // 12
    const int flat = blockIdx.x * 8 + wave;
    const int tm = flat / tilesN;                // 0..255
    const int tn = flat % tilesN;
    if (tm >= M_ / 32) return;
    const int m0 = tm * 32, n0 = tn * 64;
    const int l16 = lane & 15, lo = lane >> 4;

    v8f acc[2][4] = {};
    const bf16* pA0 = ob + (size_t)(m0 + l16) * D_ + lo * 8;
    const bf16* pA1 = pA0 + (size_t)16 * D_;

    for (int kk = 0; kk < D_; kk += 32) {
        BF16x16 a0, a1, bm[4];
        a0.h[0] = load8(pA0 + kk);  a0.h[1] = load8(pA0 + kk + 16);
        a1.h[0] = load8(pA1 + kk);  a1.h[1] = load8(pA1 + kk + 16);
#pragma unroll
        for (int c = 0; c < 4; ++c) {
            const bf16* pB = wb + (size_t)(n0 + c * 16 + l16) * D_ + kk + lo * 8;
            bm[c].h[0] = load8(pB);
            bm[c].h[1] = load8(pB + 16);
        }
#pragma unroll
        for (int c = 0; c < 4; ++c)
            acc[0][c] = __builtin_amdgcn_wmma_f32_16x16x32_bf16(
                false, a0.v, false, bm[c].v, (short)0, acc[0][c], false, false);
#pragma unroll
        for (int c = 0; c < 4; ++c)
            acc[1][c] = __builtin_amdgcn_wmma_f32_16x16x32_bf16(
                false, a1.v, false, bm[c].v, (short)0, acc[1][c], false, false);
    }

#pragma unroll
    for (int half = 0; half < 2; ++half)
#pragma unroll
        for (int c = 0; c < 4; ++c) {
            const int n = n0 + c * 16 + l16;
            const float bv = bias[n];
#pragma unroll
            for (int r = 0; r < 8; ++r) {
                const int m = m0 + half * 16 + r + lo * 8;
                out[(size_t)m * D_ + n] = acc[half][c][r] + bv;
            }
        }
}

// ------------------------------- launcher ----------------------------------
static inline size_t align256(size_t x) { return (x + 255) & ~(size_t)255; }

extern "C" void kernel_launch(void* const* d_in, const int* in_sizes, int n_in,
                              void* d_out, int out_size, void* d_ws, size_t ws_size,
                              hipStream_t stream) {
    const float* x      = (const float*)d_in[0];   // [B,T,D]
    const float* W_qkv  = (const float*)d_in[1];   // [3D,D]
    const float* b_qkv  = (const float*)d_in[2];   // [3D]
    const float* W_proj = (const float*)d_in[3];   // [D,D]
    const float* b_proj = (const float*)d_in[4];   // [D]
    float* out = (float*)d_out;                    // [B,T,D]

    char* ws = (char*)d_ws;
    const size_t nx  = (size_t)M_ * D_;            // 6.29M
    const size_t nwq = (size_t)D3_ * D_;           // 1.77M
    const size_t nwp = (size_t)D_ * D_;            // 0.59M
    const size_t nqkv = (size_t)B_ * H_ * T_ * DH_;

    bf16* xb    = (bf16*)ws;  ws += align256(nx  * sizeof(bf16));
    bf16* wqkvb = (bf16*)ws;  ws += align256(nwq * sizeof(bf16));
    bf16* wprojb= (bf16*)ws;  ws += align256(nwp * sizeof(bf16));
    bf16* Qb    = (bf16*)ws;  ws += align256(nqkv * sizeof(bf16));
    bf16* Kb    = (bf16*)ws;  ws += align256(nqkv * sizeof(bf16));
    bf16* Vt    = (bf16*)ws;  ws += align256(nqkv * sizeof(bf16));
    bf16* Ob    = (bf16*)ws;  ws += align256(nx  * sizeof(bf16));

    // 1) casts
    cvt_f32_to_bf16_kernel<<<(int)((nx  / 4 + 255) / 256), 256, 0, stream>>>(x,      xb,     (int)nx);
    cvt_f32_to_bf16_kernel<<<(int)((nwq / 4 + 255) / 256), 256, 0, stream>>>(W_qkv,  wqkvb,  (int)nwq);
    cvt_f32_to_bf16_kernel<<<(int)((nwp / 4 + 255) / 256), 256, 0, stream>>>(W_proj, wprojb, (int)nwp);

    // 2) QKV projection: (M/32)*(3D/64) = 256*36 = 9216 waves / 8 = 1152 blocks
    qkv_gemm_kernel<<<1152, 256, 0, stream>>>(xb, wqkvb, b_qkv, Qb, Kb, Vt);

    // 3) attention: B*H*(T/16) = 6144 waves / 8 = 768 blocks
    attn_kernel<<<768, 256, 0, stream>>>(Qb, Kb, Vt, Ob);

    // 4) output projection: 256*12 = 3072 waves / 8 = 384 blocks
    proj_gemm_kernel<<<384, 256, 0, stream>>>(Ob, wprojb, b_proj, out);
}
